// BestNetwork_82815559402244
// MI455X (gfx1250) — matrix operand order repaired and checked by reading, not compiled, not run on hardware
//
#include <hip/hip_runtime.h>

// ---------------------------------------------------------------------------
// BestNetwork: bidirectional gated RNN, 2 layers.
// g1-blend is the identity; both gates depend only on the layer *input*, so
// all matmuls hoist out of the recurrence as batched bf16 WMMA GEMMs, and the
// recurrence itself is 16384 independent element-wise chains of length 2048.
// GEMM weight panels are staged into LDS with the Tensor Data Mover (TDM).
// ---------------------------------------------------------------------------

static constexpr int kBS  = 32;
static constexpr int kSEQ = 2048;
static constexpr int kIN  = 512;
static constexpr int kH   = 256;
static constexpr int kM   = kBS * kSEQ;          // 65536 GEMM rows (m = b*SEQ + t)

typedef __attribute__((ext_vector_type(16))) __bf16 v16bf;
typedef __attribute__((ext_vector_type(8)))  __bf16 v8bf;
typedef __attribute__((ext_vector_type(8)))  float  v8f;
typedef __attribute__((ext_vector_type(4)))  unsigned int v4u;
typedef __attribute__((ext_vector_type(8)))  int v8i;
typedef __attribute__((ext_vector_type(4)))  int v4i;

#if __has_builtin(__builtin_amdgcn_tensor_load_to_lds)
#define HAVE_TDM 1
#endif

// ---------------------------------------------------------------------------
// TDM: DMA a [nrows x K] bf16 row-major tile from global into LDS.
// D# per CDNA5 ISA 8.3/8.4: group0 = {count=1, lds_addr, global_addr, type=2};
// group1 = {data_size=2B, tensor_dim0=K, tensor_dim1=nrows, tile_dim0=K,
// tile_dim1=nrows, tensor_dim0_stride=K}. 2-D tensor -> groups 2/3 zero.
// ---------------------------------------------------------------------------
#ifdef HAVE_TDM
__device__ __forceinline__ void tdm_load_tile_to_lds(unsigned int lds_off,
                                                     const __bf16* gsrc,
                                                     int K, int nrows)
{
    const unsigned long long ga = (unsigned long long)(uintptr_t)gsrc;
    v4u g0;
    g0[0] = 1u;                                              // count=1, no gather
    g0[1] = lds_off;                                         // LDS byte address
    g0[2] = (unsigned int)ga;                                // global_addr[31:0]
    g0[3] = (unsigned int)((ga >> 32) & 0x01FFFFFFu) | (2u << 30);  // [56:32] | type=2
    v8i g1;
    g1[0] = (int)(1u << 16);                                 // data_size = 2 bytes
    g1[1] = (int)(((unsigned)K & 0xFFFFu) << 16);            // tensor_dim0[15:0]
    g1[2] = (int)((((unsigned)K >> 16) & 0xFFFFu) |
                  (((unsigned)nrows & 0xFFFFu) << 16));      // dim0 hi | tensor_dim1 lo
    g1[3] = (int)((((unsigned)nrows >> 16) & 0xFFFFu) |
                  (((unsigned)K & 0xFFFFu) << 16));          // dim1 hi | tile_dim0=K
    g1[4] = (int)((unsigned)nrows & 0xFFFFu);                // tile_dim1 | tile_dim2=0
    g1[5] = (int)(unsigned)K;                                // tensor_dim0_stride[31:0]
    g1[6] = 0;
    g1[7] = 0;
    v4i z4 = {0, 0, 0, 0};
#if __clang_major__ >= 23
    v8i z8 = {0, 0, 0, 0, 0, 0, 0, 0};
    __builtin_amdgcn_tensor_load_to_lds(g0, g1, z4, z4, z8, 0);
#else
    __builtin_amdgcn_tensor_load_to_lds(g0, g1, z4, z4, 0);
#endif
}
#endif

// ---------------------------------------------------------------------------
// Pre-convert fp32 weights to bf16 (done once).
// ---------------------------------------------------------------------------
__global__ __launch_bounds__(256)
void cvt_f32_to_bf16(const float* __restrict__ src, __bf16* __restrict__ dst, int n)
{
    const int i = blockIdx.x * blockDim.x + threadIdx.x;
    if (i < n) dst[i] = (__bf16)src[i];
}

// ---------------------------------------------------------------------------
// GEMM: C[m,n] = sum_k A[m,k] * W[n,k] + bias[n]   (NT, both K-major)
// Block = 8 waves; wave computes 16 rows x 256 cols (16 WMMA tiles).
// Weight panel [256 x K] bf16 staged in LDS via TDM; B fragments are
// double-buffered (load tile tn+1 before WMMA of tile tn) so each WMMA
// executes under the next LDS round-trip instead of serializing on dscnt 0.
// K is a compile-time constant -> K-loop partially unrolled for a wide
// scheduling window over A global loads / B ds loads / WMMAs.
// ---------------------------------------------------------------------------
template<bool A_BF16, int K>
__global__ __launch_bounds__(256)
void gemm_bf16_wmma(const void* __restrict__ Aptr,
                    const __bf16* __restrict__ Wbf,   // [256, K] bf16
                    const float* __restrict__ bias,   // [256] fp32
                    __bf16* __restrict__ C)           // [kM, 256] bf16
{
    extern __shared__ __bf16 ldsW[];                  // [256][K]

    const int lane = threadIdx.x & 31;
    const int wave = threadIdx.x >> 5;
    const int lr   = lane & 15;                       // A row / B,C col within tile
    const int half = lane >> 4;                       // K-half selector
    const int r0   = (blockIdx.x * 8 + wave) * 16;    // M tile base

#ifdef HAVE_TDM
    if (wave == 0) {
        tdm_load_tile_to_lds((unsigned int)(uintptr_t)&ldsW[0], Wbf, K, kH);
        __builtin_amdgcn_s_wait_tensorcnt(0);
    }
#else
    for (int idx = threadIdx.x; idx < kH * K / 8; idx += 256)
        ((v8bf*)ldsW)[idx] = ((const v8bf*)Wbf)[idx];
#endif
    __syncthreads();

    v8f acc[16];
    #pragma unroll
    for (int tn = 0; tn < 16; ++tn) {
        const float bv = bias[tn * 16 + lr];          // one column per lane
        #pragma unroll
        for (int j = 0; j < 8; ++j) acc[tn][j] = bv;
    }

    const size_t arow = (size_t)(r0 + lr) * (size_t)K;

    // B fragment fetch from LDS: lane lr = column; half 0 -> K 0..7 & 16..23, half 1 -> 8..15 & 24..31
    auto loadB = [&](int tn, int k0) -> v16bf {
        const v8bf* Bv = (const v8bf*)(ldsW + (size_t)(tn * 16 + lr) * K + k0 + half * 8);
        const v8bf lo = Bv[0];
        const v8bf hi = Bv[2];
        v16bf b;
        #pragma unroll
        for (int i = 0; i < 8; ++i) { b[i] = lo[i]; b[8 + i] = hi[i]; }
        return b;
    };

    #pragma unroll 4
    for (int k0 = 0; k0 < K; k0 += 32) {
        // A fragment 16x32 (MxK): half 0 -> K k0+0..7 & 16..23, half 1 -> 8..15 & 24..31
        v16bf a;
        if (A_BF16) {
            const v8bf* Ap = (const v8bf*)((const __bf16*)Aptr + arow + k0 + half * 8);
            const v8bf lo = Ap[0];
            const v8bf hi = Ap[2];                    // +16 elements
            #pragma unroll
            for (int i = 0; i < 8; ++i) { a[i] = lo[i]; a[8 + i] = hi[i]; }
        } else {
            const float* Ap = (const float*)Aptr + arow + k0 + half * 8;
            #pragma unroll
            for (int i = 0; i < 8; ++i) { a[i] = (__bf16)Ap[i]; a[8 + i] = (__bf16)Ap[16 + i]; }
        }

        v16bf bcur = loadB(0, k0);
        #pragma unroll
        for (int tn = 0; tn < 16; ++tn) {
            v16bf bnxt = (tn < 15) ? loadB(tn + 1, k0) : bcur;   // prefetch next tile
            acc[tn] = __builtin_amdgcn_wmma_f32_16x16x32_bf16(
                false, a, false, bcur, (short)0, acc[tn], false, false);
            bcur = bnxt;
        }
    }

    // C layout: lane lr = column; VGPR j -> row r0 + half*8 + j
    #pragma unroll
    for (int tn = 0; tn < 16; ++tn) {
        const int col = tn * 16 + lr;
        #pragma unroll
        for (int j = 0; j < 8; ++j) {
            const int rr = r0 + half * 8 + j;
            C[(size_t)rr * kH + col] = (__bf16)acc[tn][j];
        }
    }
}

// ---------------------------------------------------------------------------
// Element-wise recurrence: one thread per (dir, b, n) scalar chain.
//   h = tanh(x * (sigmoid(g)*(h - x) + x))          [left == x]
// Addresses are h-independent -> unroll 8 lets the scheduler hoist the loads
// of 8 future steps above the serial transcendental chain.
// ---------------------------------------------------------------------------
__device__ __forceinline__ float fast_sigmoid(float x) {
    const float e = __builtin_amdgcn_exp2f(-1.4426950408889634f * x);
    return __builtin_amdgcn_rcpf(1.0f + e);
}
__device__ __forceinline__ float fast_tanh(float x) {
    const float e = __builtin_amdgcn_exp2f(2.8853900817779268f * x);   // exp(2x)
    return 1.0f - 2.0f * __builtin_amdgcn_rcpf(1.0f + e);              // saturates
}

__global__ __launch_bounds__(256)
void recur_dir(const __bf16* __restrict__ Xl, const __bf16* __restrict__ Xr,
               const __bf16* __restrict__ Gl, const __bf16* __restrict__ Gr,
               __bf16* __restrict__ Yl, __bf16* __restrict__ Yr,
               float* __restrict__ out_xx, float* __restrict__ hfin,
               int write_f32)
{
    const int id  = blockIdx.x * blockDim.x + threadIdx.x;   // 16384 total
    const int n   = id & (kH - 1);
    const int b   = (id >> 8) & (kBS - 1);
    const int dir = id >> 13;                                // uniform per wave

    const __bf16* X = dir ? Xr : Xl;
    const __bf16* G = dir ? Gr : Gl;
    __bf16*       Y = dir ? Yr : Yl;

    const size_t rowbase = (size_t)b * kSEQ * kH + n;
    float h = 0.0f;

    #pragma unroll 8
    for (int s = 0; s < kSEQ; ++s) {
        const int t = dir ? (kSEQ - 1 - s) : s;
        const size_t idx = rowbase + (size_t)t * kH;

        if (s + 32 < kSEQ) {   // uniform guard: cover the sequential load chain
            const size_t pidx = rowbase + (size_t)(dir ? (t - 32) : (t + 32)) * kH;
            __builtin_prefetch(X + pidx, 0, 0);
            __builtin_prefetch(G + pidx, 0, 0);
        }

        const float xv = (float)X[idx];
        const float gv = (float)G[idx];
        const float g2 = fast_sigmoid(gv);
        const float r  = fmaf(g2, h - xv, xv);   // g2*h + (1-g2)*x
        h = fast_tanh(xv * r);

        if (write_f32) {
            out_xx[((size_t)b * kSEQ + t) * (2 * kH) + (size_t)dir * kH + n] = h;
        } else {
            Y[idx] = (__bf16)h;
        }
    }
    hfin[(size_t)dir * (kBS * kH) + b * kH + n] = h;   // final state [2, bs, H]
}

// ---------------------------------------------------------------------------
// Orchestration
// ---------------------------------------------------------------------------
extern "C" void kernel_launch(void* const* d_in, const int* in_sizes, int n_in,
                              void* d_out, int out_size, void* d_ws, size_t ws_size,
                              hipStream_t stream)
{
    (void)in_sizes; (void)n_in; (void)out_size; (void)ws_size;

    const float* x    = (const float*)d_in[0];   // [32, 2048, 512]
    const float* W_fc = (const float*)d_in[1];   // [256, 512]
    const float* b_fc = (const float*)d_in[2];   // [256]
    // d_in[3]=W1, d_in[4]=b1: unused — the g1 blend is the identity.
    const float* W2   = (const float*)d_in[5];   // [2, 256, 256]
    const float* b2   = (const float*)d_in[6];   // [2, 256]

    float* out   = (float*)d_out;                         // xx: [32, 2048, 512] fp32
    float* h_out = out + (size_t)kBS * kSEQ * 2 * kH;     // [4, 32, 256] fp32

    const size_t BUF = (size_t)kM * kH;                   // 16,777,216 elems (32 MB bf16)
    __bf16* P     = (__bf16*)d_ws;     // projected input (layer-0 input, both dirs)
    __bf16* G0    = P  + BUF;          // layer-0 gate pre-activation
    __bf16* Yl    = G0 + BUF;          // layer-0 ltr output
    __bf16* Yr    = Yl + BUF;          // layer-0 rtl output (already flipped back)
    __bf16* G1l   = P;                 // reuse: layer-1 ltr gate
    __bf16* G1r   = G0;                // reuse: layer-1 rtl gate
    __bf16* WfcB  = Yr + BUF;          // bf16 weights: [256, 512]
    __bf16* W2B   = WfcB + (size_t)kH * kIN;   // [2, 256, 256]

    // 0) weights -> bf16 (once)
    cvt_f32_to_bf16<<<(kH * kIN) / 256, 256, 0, stream>>>(W_fc, WfcB, kH * kIN);
    cvt_f32_to_bf16<<<(2 * kH * kH) / 256, 256, 0, stream>>>(W2, W2B, 2 * kH * kH);

    const dim3 gblk(256);
    const dim3 ggrid(kM / 128);                       // 512 blocks, all 256 cols per block
    const size_t ldsProj  = (size_t)kH * kIN * sizeof(__bf16);   // 256 KB
    const size_t ldsLayer = (size_t)kH * kH  * sizeof(__bf16);   // 128 KB

    // 1) projection: P = x @ W_fc^T + b_fc   (K = 512)
    gemm_bf16_wmma<false, kIN><<<ggrid, gblk, ldsProj, stream>>>(x, WfcB, b_fc, P);
    // 2) layer-0 gate (shared by both directions): G0 = P @ W2[0]^T + b2[0]
    gemm_bf16_wmma<true,  kH ><<<ggrid, gblk, ldsLayer, stream>>>(P, W2B, b2, G0);
    // 3) layer-0 recurrence, both directions in one grid
    recur_dir<<<64, 256, 0, stream>>>(P, P, G0, G0, Yl, Yr, nullptr, h_out, 0);
    // 4) layer-1 gates per direction
    gemm_bf16_wmma<true,  kH ><<<ggrid, gblk, ldsLayer, stream>>>(Yl, W2B + kH * kH, b2 + kH, G1l);
    gemm_bf16_wmma<true,  kH ><<<ggrid, gblk, ldsLayer, stream>>>(Yr, W2B + kH * kH, b2 + kH, G1r);
    // 5) layer-1 recurrence -> fp32 output + final states
    recur_dir<<<64, 256, 0, stream>>>(Yl, Yr, G1l, G1r, nullptr, nullptr,
                                      out, h_out + 2 * kBS * kH, 1);
}
